// SeqToSeqGruOld_54580444397841
// MI455X (gfx1250) — matrix operand order repaired and checked by reading, not compiled, not run on hardware
//
#include <hip/hip_runtime.h>
#include <hip/hip_bf16.h>

typedef __attribute__((ext_vector_type(16))) _Float16 v16h;
typedef __attribute__((ext_vector_type(8)))  float    v8f;

// ---------------- geometry ----------------
// H=100 -> Hp=112 (7 tiles of 16 cols), K padded to 128 (4 WMMA k-steps of 32)
constexpr int MTILE  = 128;      // rows per block (8 waves x 16 rows)
constexpr int NJT    = 7;        // 112/16 output col tiles per gate
constexpr int NKS    = 4;        // 128/32 k-steps
constexpr int HSTR   = 130;      // h row stride in halves (65 dwords, gcd(65,64)=1 -> no bank conflicts)

constexpr int WHH_H  = 3 * NJT * NKS * 32 * 16;   // 43008 halves, fragment-major
constexpr int W1_H   = 4 * NKS * 32 * 16;         //  8192 halves (50->64 cols)
constexpr int WEFF_H = NJT * NKS * 32 * 16;       // 14336 halves
constexpr int PACK_H = WHH_H + W1_H + WEFF_H;     // 65536 halves = 128 KB
constexpr int HBUF_H = MTILE * HSTR;              // 16640 halves
constexpr int SMEM_HALVES = PACK_H + HBUF_H;      // 82176
constexpr int SMEM_FLOATS = 336*3 + 112 + 64 + 64 + 128; // 1376
constexpr size_t SMEM_BYTES = (size_t)SMEM_HALVES*2 + (size_t)SMEM_FLOATS*4; // 169856 B

// ---------------- fast branch-free activations ----------------
__device__ __forceinline__ float fast_sigmoid(float x) {
  // 1/(1+2^(-x*log2(e))) ; v_exp_f32 is exp2, v_rcp_f32 handles inf->0
  float e = __builtin_amdgcn_exp2f(-1.4426950408889634f * x);
  return __builtin_amdgcn_rcpf(1.f + e);
}

__device__ __forceinline__ float fast_tanh(float x) {
#if __has_builtin(__builtin_amdgcn_tanhf)
  return __builtin_amdgcn_tanhf(x);        // gfx1250 native v_tanh_f32
#else
  float e = __builtin_amdgcn_exp2f(-2.885390081777927f * x);  // 2^(-2x*log2e)
  return 2.f * __builtin_amdgcn_rcpf(1.f + e) - 1.f;
#endif
}

// ---------------- weight packers (fragment-major B layout) ----------------
// B fragment (16x16x32 f16, wave32): lane L holds column n = L%16,
// half i (0..15) holds k = ks*32 + 16*(L/16) + i. Packed contiguously per (frag,lane).

__global__ void pack_whh_k(const float* __restrict__ w_hh, _Float16* __restrict__ dst) {
  int idx = blockIdx.x * 256 + threadIdx.x;
  if (idx >= WHH_H) return;
  int i    = idx & 15;
  int lane = (idx >> 4) & 31;
  int ks   = (idx >> 9) & 3;
  int jt   = (idx >> 11) % NJT;
  int g    = idx / (NJT * 2048);
  int n = jt * 16 + (lane & 15);
  int k = ks * 32 + ((lane >> 4) << 4) + i;
  float v = (n < 100 && k < 100) ? w_hh[(g * 100 + n) * 100 + k] : 0.f;
  dst[idx] = (_Float16)v;
}

__global__ void pack_w1_k(const float* __restrict__ w1, _Float16* __restrict__ dst) {
  int idx = blockIdx.x * 256 + threadIdx.x;
  if (idx >= W1_H) return;
  int i    = idx & 15;
  int lane = (idx >> 4) & 31;
  int ks   = (idx >> 9) & 3;
  int jy   = idx >> 11;
  int n = jy * 16 + (lane & 15);
  int k = ks * 32 + ((lane >> 4) << 4) + i;
  float v = (n < 50 && k < 100) ? w1[n * 100 + k] : 0.f;
  dst[idx] = (_Float16)v;
}

// W_eff[t,h] folds the rfft into the encoder weights (FFT eliminated analytically).
__global__ void pack_weff_k(const float* __restrict__ w_enc, _Float16* __restrict__ dst) {
  int idx = blockIdx.x * 256 + threadIdx.x;
  if (idx >= WEFF_H) return;
  int i    = idx & 15;
  int lane = (idx >> 4) & 31;
  int ks   = (idx >> 9) & 3;
  int jt   = idx >> 11;
  int n = jt * 16 + (lane & 15);                 // output unit h
  int t = ks * 32 + ((lane >> 4) << 4) + i;      // time index (K)
  float acc = 0.f;
  if (n < 100 && t < 100) {
    const float* wrow = w_enc + n * 202;
    acc = wrow[t];
    for (int q = 0; q <= 50; ++q) {
      float ang = 6.283185307179586f * (float)((t * q) % 100) * 0.01f;
      float s, c;
      sincosf(ang, &s, &c);
      acc += c * wrow[100 + q] - s * wrow[151 + q];   // Re part + Im part (Im = -sin)
    }
  }
  dst[idx] = (_Float16)acc;
}

// ---------------- fragment loaders ----------------
__device__ __forceinline__ v8f wmma16(v16h a, v16h b, v8f c) {
  return __builtin_amdgcn_wmma_f32_16x16x32_f16(false, a, false, b, (short)0, c, false, false);
}

// A fragment (16x32 f16): lane L holds row M=L%16; halves 0..15 -> k=kbase..kbase+15,
// kbase = ks*32 + 16*(L/16).
__device__ __forceinline__ v16h load_a_frag(const _Float16* hrow, int kbase) {
  v16h a;
  #pragma unroll
  for (int ii = 0; ii < 8; ++ii) a[ii] = hrow[kbase + ii];
  #pragma unroll
  for (int ii = 0; ii < 8; ++ii) a[8 + ii] = hrow[kbase + 8 + ii];
  return a;
}

__device__ __forceinline__ v16h load_sig_frag(const float* sigrow, int kbase) {
  v16h a;
  #pragma unroll
  for (int ii = 0; ii < 8; ++ii) { int t = kbase + ii;      a[ii]     = (_Float16)(t < 100 ? sigrow[t] : 0.f); }
  #pragma unroll
  for (int ii = 0; ii < 8; ++ii) { int t = kbase + 8 + ii; a[8 + ii] = (_Float16)(t < 100 ? sigrow[t] : 0.f); }
  return a;
}

__device__ __forceinline__ v16h load_b_frag(const _Float16* base, int frag, int lane) {
  return *(const v16h*)(base + ((frag * 32 + lane) << 4));
}

// ---------------- fused encoder + 50-step GRU decoder ----------------
__global__ __launch_bounds__(256) void gru_fused_kernel(
    const float* __restrict__ enc,   // (N,100,1)
    const float* __restrict__ b_enc, const float* __restrict__ w_ih,
    const float* __restrict__ b_ih,  const float* __restrict__ b_hh,
    const float* __restrict__ b1,    const float* __restrict__ w2,
    const float* __restrict__ b2,
    const _Float16* __restrict__ wsw,  // packed [whh | w1 | weff]
    float* __restrict__ out)           // (N,50)
{
  extern __shared__ _Float16 smem[];
  _Float16* whh  = smem;                 // 43008 halves
  _Float16* w1p  = whh  + WHH_H;         //  8192
  _Float16* weff = w1p  + W1_H;          // 14336
  _Float16* hbuf = weff + WEFF_H;        // 16640 (128 rows x stride 130)
  float* fbase  = (float*)(hbuf + HBUF_H);
  float* wih_p  = fbase;                 // 336 (3 gates x 112, zero-padded)
  float* bih_p  = wih_p + 336;
  float* bhh_p  = bih_p + 336;
  float* benc_p = bhh_p + 336;           // 112
  float* b1_p   = benc_p + 112;          // 64
  float* w2_p   = b1_p + 64;             // 64
  float* xbuf   = w2_p + 64;             // 128 decoder scalar inputs

  const int tid  = threadIdx.x;
  const int lane = tid & 31;
  const int wave = tid >> 5;
  const int lrow = lane & 15;     // A: row-in-tile; C/D & B: column-in-tile
  const int lhi  = lane >> 4;
  const int rowbase = wave * 16;  // each wave privately owns 16 rows
  const long gbase = (long)blockIdx.x * MTILE;

  // ---- cooperative setup (only barrier in the kernel) ----
  {
    const uint4* src = (const uint4*)wsw;
    uint4* dst = (uint4*)smem;
    for (int t = tid; t < (PACK_H * 2) / 16; t += 256) dst[t] = src[t];  // 128 KB weights -> LDS
    unsigned int* hz = (unsigned int*)hbuf;
    for (int t = tid; t < HBUF_H / 2; t += 256) hz[t] = 0u;              // zero h (incl. pad cols)
    for (int t = tid; t < 336; t += 256) {
      int g = t / 112, c = t % 112;
      float wv = 0.f, bi = 0.f, bh = 0.f;
      if (c < 100) { wv = w_ih[g*100+c]; bi = b_ih[g*100+c]; bh = b_hh[g*100+c]; }
      wih_p[t] = wv; bih_p[t] = bi; bhh_p[t] = bh;
    }
    for (int t = tid; t < 112; t += 256) benc_p[t] = (t < 100) ? b_enc[t] : 0.f;
    for (int t = tid; t < 64; t += 256) {
      b1_p[t] = (t < 50) ? b1[t] : 0.f;
      w2_p[t] = (t < 50) ? w2[t] : 0.f;
    }
    for (int t = tid; t < MTILE; t += 256)
      xbuf[t] = enc[(gbase + t) * 100 + 99];   // x0 = last signal sample
  }
  __syncthreads();

  // ---- encoder: h0 = tanh(sig @ W_eff + b_enc), FFT folded into W_eff ----
  {
    const float* sigrow = enc + (gbase + rowbase + lrow) * 100;
    v16h sa[NKS];
    #pragma unroll
    for (int ks = 0; ks < NKS; ++ks) sa[ks] = load_sig_frag(sigrow, ks * 32 + (lhi << 4));
    for (int jt = 0; jt < NJT; ++jt) {
      v8f acc = {};
      #pragma unroll
      for (int ks = 0; ks < NKS; ++ks)
        acc = wmma16(sa[ks], load_b_frag(weff, jt * NKS + ks, lane), acc);
      const int c = jt * 16 + lrow;
      const float bc = benc_p[c];
      #pragma unroll
      for (int v = 0; v < 8; ++v) {
        int r = rowbase + v + (lhi << 3);
        hbuf[r * HSTR + c] = (_Float16)fast_tanh(acc[v] + bc);  // pad cols give tanh(0)=0
      }
    }
  }

  const float b2v = b2[0];
  const _Float16* hrow = hbuf + (rowbase + lrow) * HSTR;

  // ---- 50 decoder steps, h lives in LDS, no inter-wave sync needed ----
  for (int step = 0; step < 50; ++step) {
    v16h ha[NKS];
    #pragma unroll
    for (int ks = 0; ks < NKS; ++ks) ha[ks] = load_a_frag(hrow, ks * 32 + (lhi << 4));

    // gh = h @ w_hh^T for 3 gates, fused with GRU elementwise per 16-col tile
    for (int jt = 0; jt < NJT; ++jt) {
      v8f aR = {}, aZ = {}, aN = {};
      #pragma unroll
      for (int ks = 0; ks < NKS; ++ks) {
        v16h bR = load_b_frag(whh, (0 * NJT + jt) * NKS + ks, lane);
        v16h bZ = load_b_frag(whh, (1 * NJT + jt) * NKS + ks, lane);
        v16h bN = load_b_frag(whh, (2 * NJT + jt) * NKS + ks, lane);
        aR = wmma16(ha[ks], bR, aR);
        aZ = wmma16(ha[ks], bZ, aZ);
        aN = wmma16(ha[ks], bN, aN);
      }
      const int c = jt * 16 + lrow;
      const float wir = wih_p[c], wiz = wih_p[112 + c], win = wih_p[224 + c];
      const float bir = bih_p[c], biz = bih_p[112 + c], bin = bih_p[224 + c];
      const float bhr = bhh_p[c], bhz = bhh_p[112 + c], bhn = bhh_p[224 + c];
      #pragma unroll
      for (int v = 0; v < 8; ++v) {
        const int r = rowbase + v + (lhi << 3);
        const float x  = xbuf[r];
        const float rg = fast_sigmoid(x * wir + bir + aR[v] + bhr);
        const float zg = fast_sigmoid(x * wiz + biz + aZ[v] + bhz);
        const float ng = fast_tanh(x * win + bin + rg * (aN[v] + bhn));
        const float ho = (float)hbuf[r * HSTR + c];
        const float hn = (1.f - zg) * ng + zg * ho;
        hbuf[r * HSTR + c] = (_Float16)hn;      // pad cols stay exactly 0
      }
    }

    // y = leaky_relu(h_new @ w1^T + b1) @ w2^T + b2 ; feed back as next x
    #pragma unroll
    for (int ks = 0; ks < NKS; ++ks) ha[ks] = load_a_frag(hrow, ks * 32 + (lhi << 4));
    float ty[8] = {0.f, 0.f, 0.f, 0.f, 0.f, 0.f, 0.f, 0.f};
    for (int jy = 0; jy < 4; ++jy) {
      v8f acc = {};
      #pragma unroll
      for (int ks = 0; ks < NKS; ++ks)
        acc = wmma16(ha[ks], load_b_frag(w1p, jy * NKS + ks, lane), acc);
      const int c = jy * 16 + lrow;
      const float w2v = w2_p[c], b1v = b1_p[c];
      #pragma unroll
      for (int v = 0; v < 8; ++v) {
        float u = acc[v] + b1v;
        u = (u > 0.f) ? u : 0.01f * u;
        ty[v] += u * w2v;                       // padded cols contribute 0
      }
    }
    #pragma unroll
    for (int v = 0; v < 8; ++v) {
      float t = ty[v];
      t += __shfl_xor(t, 1, 32);                // reduce across the 16-lane half
      t += __shfl_xor(t, 2, 32);                // holding this row's columns
      t += __shfl_xor(t, 4, 32);
      t += __shfl_xor(t, 8, 32);
      const float y = t + b2v;
      if (lrow == 0) {
        const int r = rowbase + v + (lhi << 3);
        xbuf[r] = y;                            // same-wave LDS RAW: DScnt in-order
        out[(gbase + r) * 50 + step] = y;
      }
    }
  }
}

// ---------------- host launch ----------------
extern "C" void kernel_launch(void* const* d_in, const int* in_sizes, int n_in,
                              void* d_out, int out_size, void* d_ws, size_t ws_size,
                              hipStream_t stream) {
  const float* enc   = (const float*)d_in[0];
  const float* w_enc = (const float*)d_in[1];
  const float* b_enc = (const float*)d_in[2];
  const float* w_ih  = (const float*)d_in[3];
  const float* b_ih  = (const float*)d_in[4];
  const float* w_hh  = (const float*)d_in[5];
  const float* b_hh  = (const float*)d_in[6];
  const float* w1    = (const float*)d_in[7];
  const float* b1    = (const float*)d_in[8];
  const float* w2    = (const float*)d_in[9];
  const float* b2    = (const float*)d_in[10];
  float* out = (float*)d_out;

  _Float16* wsw     = (_Float16*)d_ws;
  _Float16* ws_whh  = wsw;
  _Float16* ws_w1   = wsw + WHH_H;
  _Float16* ws_weff = wsw + WHH_H + W1_H;

  pack_whh_k <<<(WHH_H  + 255) / 256, 256, 0, stream>>>(w_hh,  ws_whh);
  pack_w1_k  <<<(W1_H   + 255) / 256, 256, 0, stream>>>(w1,    ws_w1);
  pack_weff_k<<<(WEFF_H + 255) / 256, 256, 0, stream>>>(w_enc, ws_weff);

  const int N = in_sizes[0] / 100;   // 32768
  gru_fused_kernel<<<N / MTILE, 256, SMEM_BYTES, stream>>>(
      enc, b_enc, w_ih, b_ih, b_hh, b1, w2, b2, wsw, out);
}
